// LSTM_35081292874270
// MI455X (gfx1250) — matrix operand order repaired and checked by reading, not compiled, not run on hardware
//
#include <hip/hip_runtime.h>
#include <hip/hip_bf16.h>

// ---------------- CDNA5 WMMA vector types (probe-confirmed signatures) ------
typedef __attribute__((ext_vector_type(16))) __bf16 v16bf;
typedef __attribute__((ext_vector_type(8)))  __bf16 v8bf;
typedef __attribute__((ext_vector_type(8)))  float  v8f;
typedef __attribute__((ext_vector_type(4)))  unsigned int u32x4;
typedef __attribute__((ext_vector_type(8)))  int i32x8;
typedef __attribute__((ext_vector_type(4)))  int i32x4;

#define NB   32      // batch
#define TT   2048    // timesteps
#define DD   512     // input dim
#define HH   512     // hidden dim
#define G4H  2048    // 4*H
#define NWG  32      // persistent recurrent workgroups (16 h-cols each)

// Tensor Data Mover availability (ROCm7.2 clang-22: 5-arg builtin)
#if defined(__HIP_DEVICE_COMPILE__) && __has_builtin(__builtin_amdgcn_tensor_load_to_lds) && __has_builtin(__builtin_amdgcn_s_wait_tensorcnt)
#define HAVE_TDM 1
#else
#define HAVE_TDM 0
#endif

__device__ __forceinline__ unsigned short f2bf(float f) {
  unsigned u = __float_as_uint(f);
  u += 0x7FFFu + ((u >> 16) & 1u);           // round-to-nearest-even
  return (unsigned short)(u >> 16);
}

#if HAVE_TDM
// Generic pointers to LDS carry the byte offset in the low 32 bits.
__device__ __forceinline__ unsigned ldsof(const void* p) {
  return (unsigned)(unsigned long long)p;
}

// Build a D# and issue TENSOR_LOAD_TO_LDS for a tile of up to 3 dims.
// dsz_code: 0=1B,1=2B,2=4B. tile0 counts elements (X fastest, contiguous in LDS).
__device__ __forceinline__ void tdm_load(
    unsigned lds, const void* gptr, unsigned dsz_code,
    unsigned tile0, unsigned tile1, unsigned tile2,
    unsigned long long stride0, unsigned long long stride1,
    unsigned td0, unsigned td1, unsigned td2) {
  unsigned long long ga = (unsigned long long)gptr;
  u32x4 g0;
  g0[0] = 1u;                                        // count=1, user descriptor
  g0[1] = lds;                                       // lds_addr
  g0[2] = (unsigned)(ga & 0xFFFFFFFFu);              // global_addr[31:0]
  g0[3] = (unsigned)((ga >> 32) & 0x01FFFFFFu) | (2u << 30);  // addr[56:32] | type=2
  i32x8 g1;
  g1[0] = (int)(dsz_code << 16);                     // workgroup_mask=0 | data_size
  g1[1] = (int)((td0 & 0xFFFFu) << 16);              // tensor_dim0[15:0]
  g1[2] = (int)(((td0 >> 16) & 0xFFFFu) | ((td1 & 0xFFFFu) << 16));
  g1[3] = (int)(((td1 >> 16) & 0xFFFFu) | ((tile0 & 0xFFFFu) << 16)); // tile_dim0
  g1[4] = (int)((tile1 & 0xFFFFu) | ((tile2 & 0xFFFFu) << 16));       // tile_dim1|2
  g1[5] = (int)(stride0 & 0xFFFFFFFFull);            // tensor_dim0_stride[31:0]
  g1[6] = (int)(((stride0 >> 32) & 0xFFFFull) | ((stride1 & 0xFFFFull) << 16));
  g1[7] = (int)((stride1 >> 16) & 0xFFFFFFFFull);    // tensor_dim1_stride[47:16]
  i32x4 g2 = {(int)td2, 0, 0, 0};                    // tensor_dim2 (3D); rest 0
  i32x4 g3 = {0, 0, 0, 0};
#if __clang_major__ >= 23
  i32x8 gz8 = {0, 0, 0, 0, 0, 0, 0, 0};
  __builtin_amdgcn_tensor_load_to_lds(g0, g1, g2, g3, gz8, 0);
#else
  __builtin_amdgcn_tensor_load_to_lds(g0, g1, g2, g3, 0);
#endif
}
#endif  // HAVE_TDM

// ---------------- elementwise converts ---------------------------------------
__global__ __launch_bounds__(256) void k_cvt_bf16(const float* __restrict__ in,
                                                  unsigned short* __restrict__ out,
                                                  long long n) {
  long long i = ((long long)blockIdx.x * 256 + threadIdx.x) * 4;
  if (i + 3 < n) {
    float4 v = *reinterpret_cast<const float4*>(in + i);
    out[i + 0] = f2bf(v.x); out[i + 1] = f2bf(v.y);
    out[i + 2] = f2bf(v.z); out[i + 3] = f2bf(v.w);
  }
}

// W [512][2048] f32 -> Wt [2048][512] bf16 (weights column-major so WMMA
// B-fragments are contiguous 32B runs along K)
__global__ __launch_bounds__(256) void k_transpose_bf16(const float* __restrict__ w,
                                                        unsigned short* __restrict__ wT) {
  long long idx = (long long)blockIdx.x * 256 + threadIdx.x;
  int c = (int)(idx >> 9);
  int k = (int)(idx & 511);
  wT[idx] = f2bf(w[(long long)k * G4H + c]);
}

__global__ __launch_bounds__(256) void k_init(const float* __restrict__ h0,
                                              unsigned short* __restrict__ hbuf0,
                                              unsigned* __restrict__ bar) {
  int i = blockIdx.x * 256 + threadIdx.x;
  if (i < NB * HH) hbuf0[i] = f2bf(h0[i]);
  if (i == 0) bar[0] = 0u;                    // reset grid barrier every launch
}

// ---------------- Phase 1: affine = x @ Wx + b  (bf16 WMMA, f32 out) ---------
// 128x128 C-tile per 256-thread block; K staged 32 at a time through
// double-buffered LDS fed by the Tensor Data Mover (async, TENSORcnt).
__global__ __launch_bounds__(256) void k_gemm_affine(
    const unsigned short* __restrict__ xb,    // [65536][512] bf16
    const unsigned short* __restrict__ wxT,   // [2048][512]  bf16
    const float* __restrict__ bias,           // [2048]
    float* __restrict__ affine) {             // [T][NB][4H]  f32
#if HAVE_TDM
  __shared__ __align__(16) unsigned short As[2][128][32];
  __shared__ __align__(16) unsigned short Bs[2][128][32];
#else
  __shared__ __align__(16) unsigned short As[1][128][32];
  __shared__ __align__(16) unsigned short Bs[1][128][32];
#endif
  const int tid  = threadIdx.x;
  const int wave = tid >> 5;
  const int lane = tid & 31;
  const int hf   = lane >> 4;        // lane-half selects K sub-range
  const int n    = lane & 15;        // M/N index within 16
  const int rowbase = blockIdx.x * 128;
  const int colbase = blockIdx.y * 128;

#if HAVE_TDM
  if (tid == 0) {   // prologue: DMA k-slice 0 into buffer 0
    tdm_load(ldsof(&As[0][0][0]), xb  + (long long)rowbase * DD, 1,
             32, 128, 0, DD, 0, 32, 128, 0);
    tdm_load(ldsof(&Bs[0][0][0]), wxT + (long long)colbase * DD, 1,
             32, 128, 0, DD, 0, 32, 128, 0);
  }
#endif

  v8f acc[8] = {};
  for (int ki = 0; ki < 16; ++ki) {
    const int kb = ki * 32;
#if HAVE_TDM
    const int cur = ki & 1;
    if (tid == 0) {
      if (ki + 1 < 16) {   // DMA next slice while we compute this one
        tdm_load(ldsof(&As[cur ^ 1][0][0]), xb  + (long long)rowbase * DD + kb + 32, 1,
                 32, 128, 0, DD, 0, 32, 128, 0);
        tdm_load(ldsof(&Bs[cur ^ 1][0][0]), wxT + (long long)colbase * DD + kb + 32, 1,
                 32, 128, 0, DD, 0, 32, 128, 0);
        __builtin_amdgcn_s_wait_tensorcnt(2);   // slice ki has landed
      } else {
        __builtin_amdgcn_s_wait_tensorcnt(0);
      }
    }
    __syncthreads();
    const unsigned short (*Ac)[32] = As[cur];
    const unsigned short (*Bc)[32] = Bs[cur];
#else
    {
      const int r2 = tid >> 1, part = tid & 1;
      *reinterpret_cast<float4*>(&As[0][r2][part * 16]) =
          *reinterpret_cast<const float4*>(&xb[(long long)(rowbase + r2) * DD + kb + part * 16]);
      *reinterpret_cast<float4*>(&Bs[0][r2][part * 16]) =
          *reinterpret_cast<const float4*>(&wxT[(long long)(colbase + r2) * DD + kb + part * 16]);
    }
    __syncthreads();
    const unsigned short (*Ac)[32] = As[0];
    const unsigned short (*Bc)[32] = Bs[0];
#endif

    // A frag (16x32 bf16): lanes 0-15 K=0..15, lanes 16-31 K=16..31
    union { v16bf v; v8bf h[2]; } a;
    const unsigned short* ar = &Ac[wave * 16 + n][0];
    a.h[0] = *reinterpret_cast<const v8bf*>(ar + 16 * hf);
    a.h[1] = *reinterpret_cast<const v8bf*>(ar + 16 * hf + 8);

    // batch-load all 8 B frags first so one dscnt wait covers 8 WMMAs
    v16bf bq[8];
#pragma unroll
    for (int nt = 0; nt < 8; ++nt)
      bq[nt] = *reinterpret_cast<const v16bf*>(&Bc[nt * 16 + n][16 * hf]);
#pragma unroll
    for (int nt = 0; nt < 8; ++nt)
      acc[nt] = __builtin_amdgcn_wmma_f32_16x16x32_bf16(
          false, a.v, false, bq[nt], (short)0, acc[nt], false, false);
    __syncthreads();
  }

#pragma unroll
  for (int nt = 0; nt < 8; ++nt) {
    const int col = colbase + nt * 16 + n;
    const float bv = bias[col];
#pragma unroll
    for (int r = 0; r < 8; ++r) {                 // C frag: M = r + 8*hf
      const int rowg = rowbase + wave * 16 + r + 8 * hf;   // = bat*T + t
      const int bat = rowg >> 11;
      const int t   = rowg & 2047;
      affine[((long long)t * NB + bat) * G4H + col] = acc[nt][r] + bv;
    }
  }
}

// ---------------- Phase 2: persistent recurrent LSTM kernel ------------------
// 32 WGs; WG j owns h columns [j*16, j*16+16). Wh^T slice TDM'd into LDS once;
// the per-step affine slab is TDM-prefetched (3D gather) one step ahead.
__global__ __launch_bounds__(256) void k_lstm(
    const float* __restrict__ affine,           // [T][NB][4H] f32
    const unsigned short* __restrict__ whT_g,   // [2048][512] bf16
    unsigned short* __restrict__ hbufs,         // 2 x [NB][HH] bf16
    float* __restrict__ out,                    // [NB][T][HH] f32
    unsigned* __restrict__ bar) {
  __shared__ __align__(16) unsigned short Ws[64][512];   // Wh^T slice, 64KB
  __shared__ __align__(16) unsigned short hl[32][512];   // h(t) bf16, 32KB
  __shared__ float gates[4][32][16];                     // pre-activations, 8KB
  __shared__ float cst[32][16];                          // cell state, 2KB
#if HAVE_TDM
  __shared__ __align__(16) float affl[2][32][4][16];     // affine slabs, 16KB
#endif

  const int tid  = threadIdx.x;
  const int wave = tid >> 5;
  const int lane = tid & 31;
  const int hf   = lane >> 4;
  const int n    = lane & 15;
  const int j    = blockIdx.x;       // 0..31 : h-column chunk
  const int mt   = wave >> 2;        // batch tile 0..1
  const int nt   = wave & 3;         // gate 0..3 (i,f,o,g)

#if HAVE_TDM
  if (tid == 0) {
    // Wh^T slice: 4 gate chunks of 16 rows x 512 K, each one 2D TDM tile
#pragma unroll
    for (int g = 0; g < 4; ++g)
      tdm_load(ldsof(&Ws[g * 16][0]),
               whT_g + (long long)(g * 512 + j * 16) * 512, 1,
               512, 16, 0, 512, 0, 512, 16, 0);
    // affine slab for t=0: 3D tile x=16 cols, y=4 gates (stride 512),
    // z=32 batch (stride 2048)
    tdm_load(ldsof(&affl[0][0][0][0]), affine + j * 16, 2,
             16, 4, 32, 512, 2048, 16, 4, 32);
    __builtin_amdgcn_s_wait_tensorcnt(1);   // Ws loads done; affl#0 may fly on
  }
#else
  for (int ch = tid; ch < 4096; ch += 256) {
    const int row = ch >> 6;
    const int off = (ch & 63) << 3;
    const int g = row >> 4, cl = row & 15;
    *reinterpret_cast<float4*>(&Ws[row][off]) =
        *reinterpret_cast<const float4*>(&whT_g[((long long)(g * 512 + j * 16 + cl)) * 512 + off]);
  }
#endif
  for (int e = tid; e < 512; e += 256) cst[e >> 4][e & 15] = 0.0f;
  __syncthreads();

#pragma unroll 1
  for (int t = 0; t < TT; ++t) {
    const unsigned short* hg = hbufs + (long long)(t & 1) * (NB * HH);
    unsigned short*       hn = hbufs + (long long)((t + 1) & 1) * (NB * HH);

#if HAVE_TDM
    // kick off next step's affine slab; overlaps h-stage + WMMA below
    if (tid == 0 && t + 1 < TT)
      tdm_load(ldsof(&affl[(t + 1) & 1][0][0][0]),
               affine + (long long)(t + 1) * (NB * G4H) + j * 16, 2,
               16, 4, 32, 512, 2048, 16, 4, 32);
#endif

    // stage h(t) (32KB) into LDS — shared by all 8 waves' A-fragments
    for (int ch = tid; ch < 2048; ch += 256) {
      *(reinterpret_cast<float4*>(&hl[0][0]) + ch) =
          *(reinterpret_cast<const float4*>(hg) + ch);
    }
    __syncthreads();

    // 16x16 gate subtile = h(batch tile) @ Wh(gate cols); K=512 in 16 WMMAs.
    // Fragments loaded in blocks of 4 so one wait covers 4 WMMAs; 4 split
    // accumulators break the WMMA->WMMA RAW chain.
    v8f acc[4] = {};
    const unsigned short* arow = &hl[mt * 16 + n][0];
    const unsigned short* wrow = &Ws[nt * 16 + n][0];
#pragma unroll
    for (int kq = 0; kq < 4; ++kq) {
      union { v16bf v; v8bf h[2]; } a[4];
      v16bf bq[4];
#pragma unroll
      for (int u = 0; u < 4; ++u) {
        const int kb = (kq * 4 + u) * 32;
        a[u].h[0] = *reinterpret_cast<const v8bf*>(arow + kb + 16 * hf);
        a[u].h[1] = *reinterpret_cast<const v8bf*>(arow + kb + 16 * hf + 8);
        bq[u] = *reinterpret_cast<const v16bf*>(wrow + kb + 16 * hf);
      }
#pragma unroll
      for (int u = 0; u < 4; ++u)
        acc[u] = __builtin_amdgcn_wmma_f32_16x16x32_bf16(
            false, a[u].v, false, bq[u], (short)0, acc[u], false, false);
    }
    v8f acct = (acc[0] + acc[1]) + (acc[2] + acc[3]);

#if HAVE_TDM
    if (tid == 0) {          // make sure this step's affine slab has landed
      if (t + 1 < TT) __builtin_amdgcn_s_wait_tensorcnt(1);
      else            __builtin_amdgcn_s_wait_tensorcnt(0);
    }
    __syncthreads();
#endif

    // exchange gate pre-activations (+= affine_x) through LDS
#pragma unroll
    for (int r = 0; r < 8; ++r) {
      const int bb = mt * 16 + r + 8 * hf;
#if HAVE_TDM
      gates[nt][bb][n] = acct[r] + affl[t & 1][bb][nt][n];
#else
      const int gcol = nt * 512 + j * 16 + n;
      const float* aff_t = affine + (long long)t * (NB * G4H);
      gates[nt][bb][n] = acct[r] + aff_t[(long long)bb * G4H + gcol];
      if (t + 1 < TT)
        __builtin_prefetch(aff_t + NB * G4H + (long long)bb * G4H + gcol, 0, 1);
#endif
    }
    __syncthreads();

    // elementwise LSTM cell: 512 elems, 2 per thread
#pragma unroll
    for (int e = tid; e < 512; e += 256) {
      const int bb = e >> 4, cc = e & 15;
      const float iv = 1.0f / (1.0f + __expf(-gates[0][bb][cc]));
      const float fv = 1.0f / (1.0f + __expf(-gates[1][bb][cc]));
      const float ov = 1.0f / (1.0f + __expf(-gates[2][bb][cc]));
      const float gv = tanhf(gates[3][bb][cc]);
      const float cn = fv * cst[bb][cc] + iv * gv;
      cst[bb][cc] = cn;
      const float hv = ov * tanhf(cn);
      out[((long long)bb * TT + t) * HH + j * 16 + cc] = hv;
      hn[(long long)bb * HH + j * 16 + cc] = f2bf(hv);
    }

    // ---- one grid sync per step (double-buffered h avoids a second one) ----
    __threadfence();        // release: h(t+1) stores visible device-wide
    __syncthreads();
    if (tid == 0) {
      __builtin_amdgcn_s_cluster_barrier();   // HW fast path if clustered
      atomicAdd(bar, 1u);
      const unsigned target = (unsigned)(t + 1) * (unsigned)NWG;
      while (atomicAdd(bar, 0u) < target) { __builtin_amdgcn_s_sleep(2); }
    }
    __syncthreads();
    __threadfence();        // acquire: don't read stale h through WGP$
  }
}

// ---------------- launcher ----------------------------------------------------
extern "C" void kernel_launch(void* const* d_in, const int* in_sizes, int n_in,
                              void* d_out, int out_size, void* d_ws, size_t ws_size,
                              hipStream_t stream) {
  (void)in_sizes; (void)n_in; (void)out_size; (void)ws_size;
  const float* x  = (const float*)d_in[0];   // [32][2048][512]
  const float* h0 = (const float*)d_in[1];   // [32][512]
  const float* Wx = (const float*)d_in[2];   // [512][2048]
  const float* Wh = (const float*)d_in[3];   // [512][2048]
  const float* b  = (const float*)d_in[4];   // [2048]
  float* out = (float*)d_out;                // [32][2048][512]

  char* ws = (char*)d_ws;
  const unsigned long long SZ_AFF = (unsigned long long)TT * NB * G4H * 4ull; // 512MB
  const unsigned long long SZ_XB  = (unsigned long long)NB * TT * DD * 2ull;  // 64MB
  const unsigned long long SZ_WT  = (unsigned long long)G4H * DD * 2ull;      // 2MB each
  const unsigned long long SZ_HB  = 2ull * NB * HH * 2ull;                    // 64KB

  float*          affine = (float*)(ws);
  unsigned short* xb     = (unsigned short*)(ws + SZ_AFF);
  unsigned short* wxT    = (unsigned short*)(ws + SZ_AFF + SZ_XB);
  unsigned short* whT    = (unsigned short*)(ws + SZ_AFF + SZ_XB + SZ_WT);
  unsigned short* hbufs  = (unsigned short*)(ws + SZ_AFF + SZ_XB + 2 * SZ_WT);
  unsigned*       bar    = (unsigned*)(ws + SZ_AFF + SZ_XB + 2 * SZ_WT + SZ_HB);

  const long long nx = (long long)NB * TT * DD;              // 33,554,432
  k_cvt_bf16<<<(unsigned)(nx / 4 / 256), 256, 0, stream>>>(x, xb, nx);
  k_transpose_bf16<<<(G4H * DD) / 256, 256, 0, stream>>>(Wx, wxT);
  k_transpose_bf16<<<(G4H * DD) / 256, 256, 0, stream>>>(Wh, whT);
  k_init<<<64, 256, 0, stream>>>(h0, hbufs, bar);
  k_gemm_affine<<<dim3((NB * TT) / 128, G4H / 128), 256, 0, stream>>>(xb, wxT, b, affine);
  k_lstm<<<NWG, 256, 0, stream>>>(affine, whT, hbufs, out, bar);
}
